// MaskedLinear_37915971289107
// MI455X (gfx1250) — matrix operand order repaired and checked by reading, not compiled, not run on hardware
//
#include <hip/hip_runtime.h>
#include <stdint.h>

// ---------------------------------------------------------------------------
// MaskedLinear: y = x @ W^T + b, rows zeroed where !amask
//   x: [N,128] f32, amask: [N] bool(1B), W: [128,128] f32, b: [128] f32
//   out: [N,128] f32
// Memory-bound (~1.03 GB @ 23.3 TB/s => ~44us floor). bf16 WMMA
// (V_WMMA_F32_16X16X32_BF16, f32 accum) keeps compute far off the roofline.
// This revision: wave-uniform (readfirstlane) store-path selection to kill
// per-element exec-mask juggling, and pure 32-bit index arithmetic so stores
// are saddr/ioffset-form instead of per-element 64-bit VALU adds.
// ---------------------------------------------------------------------------

typedef __attribute__((ext_vector_type(16))) __bf16          v16bf;
typedef __attribute__((ext_vector_type(16))) unsigned short  v16us;
typedef __attribute__((ext_vector_type(8)))  float           v8f;

#define IN_F  128
#define OUT_F 128

// Pack 16 f32 -> 16 bf16 via native converts (lowers to v_cvt_pk_bf16_f32).
__device__ __forceinline__ v16bf cvt16(const float4 a0, const float4 a1,
                                       const float4 a2, const float4 a3) {
    v16bf u;
    u[0]  = (__bf16)a0.x; u[1]  = (__bf16)a0.y; u[2]  = (__bf16)a0.z; u[3]  = (__bf16)a0.w;
    u[4]  = (__bf16)a1.x; u[5]  = (__bf16)a1.y; u[6]  = (__bf16)a1.z; u[7]  = (__bf16)a1.w;
    u[8]  = (__bf16)a2.x; u[9]  = (__bf16)a2.y; u[10] = (__bf16)a2.z; u[11] = (__bf16)a2.w;
    u[12] = (__bf16)a3.x; u[13] = (__bf16)a3.y; u[14] = (__bf16)a3.z; u[15] = (__bf16)a3.w;
    return u;
}

// Load the 4 K-chunk A-fragments for one 16-row tile.
// 16-bit A 16x32 layout (contiguous K, matching B): lane L<16 holds row L with
//   VGPR0..7 = K=k0..k0+15 ; lanes 16-31: +16.
__device__ __forceinline__ void load_afrags(const float* __restrict__ xr,
                                            const int hi, v16bf frag[4]) {
    #pragma unroll
    for (int kc = 0; kc < 4; ++kc) {
        const int k0 = kc * 32 + hi * 16;
        const float4 q0 = *(const float4*)(xr + k0);
        const float4 q1 = *(const float4*)(xr + k0 + 4);
        const float4 q2 = *(const float4*)(xr + k0 + 8);
        const float4 q3 = *(const float4*)(xr + k0 + 12);
        frag[kc] = cvt16(q0, q1, q2, q3);
    }
}

// Per-row mask scales for one tile (C/D layout: VGPR r -> row r + hi*8).
__device__ __forceinline__ void load_scales(const unsigned char* __restrict__ amask,
                                            uint32_t tileR, int hi, uint32_t nrows,
                                            float scale[8]) {
    uint32_t mb = tileR + (uint32_t)(hi * 8);
    if (mb + 8u > nrows) mb = (nrows >= 8u) ? nrows - 8u : 0u;
    uint32_t mlo, mhi; // two 32-bit halves: avoids 64-bit compare patterns
    __builtin_memcpy(&mlo, amask + mb, 4);
    __builtin_memcpy(&mhi, amask + mb + 4, 4);
    #pragma unroll
    for (int r = 0; r < 4; ++r) {
        scale[r]     = ((mlo >> (8 * r)) & 0xFFu) ? 1.0f : 0.0f;
        scale[r + 4] = ((mhi >> (8 * r)) & 0xFFu) ? 1.0f : 0.0f;
    }
}

__global__ __launch_bounds__(256)
void masked_linear_wmma_kernel(const float* __restrict__ x,
                               const unsigned char* __restrict__ amask,
                               const float* __restrict__ W,
                               const float* __restrict__ bias,
                               float* __restrict__ out,
                               int nrows_i)
{
    __shared__ __align__(32) unsigned short sW[OUT_F * IN_F]; // W as bf16, 32 KB
    __shared__ float sB[OUT_F];

    const uint32_t nrows = (uint32_t)nrows_i;
    const int tid = threadIdx.x;

    // ---- Stage W (f32 -> bf16) into LDS: coalesced float4 loads ----
    for (int i = tid; i < (OUT_F * IN_F) / 4; i += 256) {
        const float4 v = ((const float4*)W)[i];
        const unsigned short p0 = __builtin_bit_cast(unsigned short, (__bf16)v.x);
        const unsigned short p1 = __builtin_bit_cast(unsigned short, (__bf16)v.y);
        const unsigned short p2 = __builtin_bit_cast(unsigned short, (__bf16)v.z);
        const unsigned short p3 = __builtin_bit_cast(unsigned short, (__bf16)v.w);
        uint2 pk;
        pk.x = (uint32_t)p0 | ((uint32_t)p1 << 16);
        pk.y = (uint32_t)p2 | ((uint32_t)p3 << 16);
        ((uint2*)sW)[i] = pk;
    }
    if (tid < OUT_F) sB[tid] = bias[tid];
    __syncthreads();

    const int wave = tid >> 5;
    const int lane = tid & 31;
    const int hi   = lane >> 4;    // 0: lanes 0-15, 1: lanes 16-31
    const int lrow = lane & 15;

    // Each wave owns two 16-row tiles (32 rows); block covers 256 rows.
    const uint32_t R0 = (uint32_t)blockIdx.x * 256u + (uint32_t)wave * 32u;
    // Wave-uniform by construction; force scalar branch (no exec masking).
    if (__builtin_amdgcn_readfirstlane((int)(R0 >= nrows))) return;
    const uint32_t R1 = R0 + 16u;

    // ---- A fragments for both tiles (loads row-clamped; stores guarded) ----
    uint32_t row0 = R0 + (uint32_t)lrow;
    uint32_t row1 = R1 + (uint32_t)lrow;
    if (row0 >= nrows) row0 = nrows - 1u;
    if (row1 >= nrows) row1 = nrows - 1u;

    v16bf a0[4], a1[4];
    load_afrags(x + row0 * (uint32_t)IN_F, hi, a0);
    load_afrags(x + row1 * (uint32_t)IN_F, hi, a1);

    float scale0[8], scale1[8];
    load_scales(amask, R0, hi, nrows, scale0);
    load_scales(amask, R1, hi, nrows, scale1);

    // Wave-uniform "all 32 rows in range" fast-path flag, in an SGPR.
    const int fullS = __builtin_amdgcn_readfirstlane((int)((R0 + 32u) <= nrows));

    // Store bases (32-bit element indices; max ~1.28e8 << 2^31).
    const uint32_t so0 = (R0 + (uint32_t)(hi * 8)) * (uint32_t)OUT_F + (uint32_t)lrow;
    const uint32_t so1 = so0 + 16u * (uint32_t)OUT_F;

    // ---- 8 output tiles of 16 columns; K reduced in 4 chunks of 32 ----
    #pragma unroll
    for (int nt = 0; nt < 8; ++nt) {
        const int ob = nt * 16;
        v8f acc0 = {};
        v8f acc1 = {};
        #pragma unroll
        for (int kc = 0; kc < 4; ++kc) {
            // 16-bit B 32x16, contiguous K as in A: column = lane&15; lanes>=16 take K+16.
            const int kg = kc * 32 + hi * 16;
            const v16us braw = *(const v16us*)&sW[(ob + lrow) * IN_F + kg];
            const v16bf bfrag = __builtin_bit_cast(v16bf, braw);
            acc0 = __builtin_amdgcn_wmma_f32_16x16x32_bf16(
                       false, a0[kc], false, bfrag, (short)0, acc0, false, false);
            acc1 = __builtin_amdgcn_wmma_f32_16x16x32_bf16(
                       false, a1[kc], false, bfrag, (short)0, acc1, false, false);
        }
        // Epilogue: (acc + b) * mask.
        const float bval = sB[ob + lrow];
        if (fullS) {
            // Unpredicated: single base pointer per tile, const ioffset stores.
            float* __restrict__ p0 = out + (so0 + (uint32_t)ob);
            float* __restrict__ p1 = out + (so1 + (uint32_t)ob);
            #pragma unroll
            for (int r = 0; r < 8; ++r)
                p0[r * OUT_F] = (acc0[r] + bval) * scale0[r];
            #pragma unroll
            for (int r = 0; r < 8; ++r)
                p1[r * OUT_F] = (acc1[r] + bval) * scale1[r];
        } else {
            // Rare tail tile: per-row guards.
            float* __restrict__ p0 = out + (so0 + (uint32_t)ob);
            float* __restrict__ p1 = out + (so1 + (uint32_t)ob);
            #pragma unroll
            for (int r = 0; r < 8; ++r) {
                if (R0 + (uint32_t)(hi * 8) + (uint32_t)r < nrows)
                    p0[r * OUT_F] = (acc0[r] + bval) * scale0[r];
            }
            #pragma unroll
            for (int r = 0; r < 8; ++r) {
                if (R1 + (uint32_t)(hi * 8) + (uint32_t)r < nrows)
                    p1[r * OUT_F] = (acc1[r] + bval) * scale1[r];
            }
        }
    }
}

extern "C" void kernel_launch(void* const* d_in, const int* in_sizes, int n_in,
                              void* d_out, int out_size, void* d_ws, size_t ws_size,
                              hipStream_t stream) {
    const float*         xp = (const float*)d_in[0];
    const unsigned char* mp = (const unsigned char*)d_in[1]; // numpy bool: 1 byte
    const float*         Wp = (const float*)d_in[2];
    const float*         bp = (const float*)d_in[3];
    float*               op = (float*)d_out;

    const int nrows  = in_sizes[1];              // N from amask
    const int blocks = (nrows + 255) / 256;      // 256 rows per block

    masked_linear_wmma_kernel<<<blocks, 256, 0, stream>>>(xp, mp, Wp, bp, op, nrows);
}
